// CausalSelfAttention_43705587204776
// MI455X (gfx1250) — matrix operand order repaired and checked by reading, compiled, not run on hardware
//
#include <hip/hip_runtime.h>

#define DEVI __device__ __forceinline__

typedef __attribute__((ext_vector_type(16))) _Float16 v16h;
typedef __attribute__((ext_vector_type(8)))  _Float16 v8h;
typedef __attribute__((ext_vector_type(8)))  float    v8f;

constexpr int kB  = 2;
constexpr int kN  = 2048;
constexpr int kD  = 1024;
constexpr int kH  = 16;
constexpr int kDH = 64;
constexpr int kE  = 192;   // 3*DH  (chunk order: k, q, v)

// ---------------------------------------------------------------------------
// WMMA fragment loaders (wave32, V_WMMA_F32_16X16X32_F16)
// A (M=16 x K=32): lane l holds row m=l&15; elems 0..15 = K kb..kb+15
//   contiguous, kb = 16*(l>>4).                     (matches B order)
// B (K=32 x N=16): lane l holds column n=l&15; elems 0..15 = K kb2..kb2+15
//   contiguous, kb2 = 16*(l>>4).                    (matches A order)
// C/D f32: VGPR r, lanes 0-15 -> row r, lanes 16-31 -> row r+8; col = l&15.
// ---------------------------------------------------------------------------
DEVI v16h load_fragA(const _Float16* base, int stride, int lane) {
  const int r  = lane & 15;
  const int kb = (lane >> 4) << 4;
  const _Float16* p = base + r * stride + kb;
  return *(const v16h*)(p);
}

// base = B^T stored row-major (row n of base = column n of B), contiguous in K
DEVI v16h load_fragB(const _Float16* base, int stride, int lane) {
  const int n  = lane & 15;
  const int kb = (lane >> 4) << 4;
  return *(const v16h*)(base + n * stride + kb);
}

DEVI v8f wmma_f16(v16h a, v16h b, v8f c) {
  return __builtin_amdgcn_wmma_f32_16x16x32_f16(false, a, false, b, (short)0, c,
                                                false, false);
}

// ---------------------------------------------------------------------------
// Stage 0: conversions / packing
// ---------------------------------------------------------------------------
__global__ void cvt_f32_to_f16(const float* __restrict__ in,
                               _Float16* __restrict__ out, int n) {
  int i = blockIdx.x * blockDim.x + threadIdx.x;
  int st = gridDim.x * blockDim.x;
  for (; i < n; i += st) out[i] = (_Float16)in[i];
}

// W_kqv [H, D, 192] -> Wt [H, 192, D]  (f16, transposed: B-frags contiguous)
__global__ void cvt_wkqv_t(const float* __restrict__ W,
                           _Float16* __restrict__ Wt) {
  int n = kH * kD * kE;
  int i = blockIdx.x * blockDim.x + threadIdx.x;
  int st = gridDim.x * blockDim.x;
  for (; i < n; i += st) {
    int h   = i / (kD * kE);
    int rem = i % (kD * kE);
    int d   = rem / kE;
    int e   = rem % kE;
    Wt[((size_t)h * kE + e) * kD + d] = (_Float16)W[i];
  }
}

// ---------------------------------------------------------------------------
// Stage 1: KQV projection.  One wave -> 16 rows x 64 cols (one k/q/v chunk).
// grid (N/16, 3, B*H), block 32.  blockIdx.y: 0=k, 1=q(scaled), 2=v(transposed)
// Software-pipelined: next k-step's fragments are in flight during WMMAs.
// ---------------------------------------------------------------------------
__global__ void kqv_gemm(const _Float16* __restrict__ xh,
                         const _Float16* __restrict__ Wt,
                         const float* __restrict__ bias_kqv,
                         _Float16* __restrict__ qbuf,
                         _Float16* __restrict__ kbuf,
                         _Float16* __restrict__ vt) {
  const int lane = threadIdx.x & 31;
  const int hi   = lane >> 4;
  const int ln   = lane & 15;
  const int bh = blockIdx.z, b = bh >> 4, h = bh & 15;
  const int row0  = blockIdx.x * 16;
  const int chunk = blockIdx.y;                // 0=k 1=q 2=v
  const int e0    = chunk * 64;

  const _Float16* xrow  = xh + ((size_t)b * kN + row0) * kD;
  const _Float16* wbase = Wt + ((size_t)h * kE + e0) * kD;

  v8f acc0 = {}, acc1 = {}, acc2 = {}, acc3 = {};
  v16h a  = load_fragA(xrow, kD, lane);
  v16h b0 = load_fragB(wbase,               kD, lane);
  v16h b1 = load_fragB(wbase + 16 * kD,     kD, lane);
  v16h b2 = load_fragB(wbase + 32 * kD,     kD, lane);
  v16h b3 = load_fragB(wbase + 48 * kD,     kD, lane);

#pragma unroll 1
  for (int kk = 0; kk < kD / 32 - 1; ++kk) {
    const int ko = (kk + 1) * 32;
    v16h an  = load_fragA(xrow + ko, kD, lane);
    v16h bn0 = load_fragB(wbase + ko,           kD, lane);
    v16h bn1 = load_fragB(wbase + 16 * kD + ko, kD, lane);
    v16h bn2 = load_fragB(wbase + 32 * kD + ko, kD, lane);
    v16h bn3 = load_fragB(wbase + 48 * kD + ko, kD, lane);
    acc0 = wmma_f16(a, b0, acc0);
    acc1 = wmma_f16(a, b1, acc1);
    acc2 = wmma_f16(a, b2, acc2);
    acc3 = wmma_f16(a, b3, acc3);
    a = an; b0 = bn0; b1 = bn1; b2 = bn2; b3 = bn3;
  }
  acc0 = wmma_f16(a, b0, acc0);
  acc1 = wmma_f16(a, b1, acc1);
  acc2 = wmma_f16(a, b2, acc2);
  acc3 = wmma_f16(a, b3, acc3);
  v8f acc[4] = {acc0, acc1, acc2, acc3};

  const size_t head = (size_t)(b * kH + h);
  if (chunk == 2) {
    // v: store transposed [B,H,DH,N]; 8 consecutive rows per lane -> one b128
#pragma unroll
    for (int dt = 0; dt < 4; ++dt) {
      int d = dt * 16 + ln;
      float bv = bias_kqv[h * kE + 128 + d];
      v8h hv;
#pragma unroll
      for (int r = 0; r < 8; ++r) hv[r] = (_Float16)(acc[dt][r] + bv);
      *(v8h*)(vt + (head * kDH + d) * kN + row0 + 8 * hi) = hv;
    }
  } else {
    _Float16* dst = (chunk == 0) ? kbuf : qbuf;
    const float scale = (chunk == 0) ? 1.0f : 0.125f;  // 1/sqrt(64) folded into q
#pragma unroll
    for (int dt = 0; dt < 4; ++dt) {
      int e = dt * 16 + ln;
      float bv = bias_kqv[h * kE + e0 + e];
#pragma unroll
      for (int r = 0; r < 8; ++r) {
        int row = row0 + r + 8 * hi;
        dst[(head * kN + row) * kDH + e] = (_Float16)((acc[dt][r] + bv) * scale);
      }
    }
  }
}

// ---------------------------------------------------------------------------
// Stage 2: streaming causal attention. 1 wave per 16 query rows.
// grid (N/16, B*H), block 32.  V is pre-transposed [B,H,DH,N].
// ---------------------------------------------------------------------------
__global__ void flash_attn(const _Float16* __restrict__ qbuf,
                           const _Float16* __restrict__ kbuf,
                           const _Float16* __restrict__ vt,
                           _Float16* __restrict__ sabuf) {
  __shared__ float    Sld[16][32];
  __shared__ _Float16 Pld[16][32];
  __shared__ float    Ald[16];
  __shared__ float    Lld[16];

  const int lane = threadIdx.x & 31;
  const int hi   = lane >> 4;
  const int ln   = lane & 15;
  const int bh = blockIdx.y, b = bh >> 4, h = bh & 15;
  const int r0 = blockIdx.x * 16;
  const size_t head = (size_t)(b * kH + h);

  const _Float16* qp = qbuf + (head * kN + r0) * kDH;
  v16h qa0 = load_fragA(qp,      kDH, lane);   // d 0..31
  v16h qa1 = load_fragA(qp + 32, kDH, lane);   // d 32..63

  v8f o[4] = {v8f{}, v8f{}, v8f{}, v8f{}};
  float Mrow = -1e30f, Lrow = 0.0f;            // duplicated in lanes l and l+16

  const int row  = ln;                          // softmax row owned by this lane
  const int coff = hi * 16;                     // column half
  const int g    = r0 + row;

  const int ntiles = (r0 + 15) / 32 + 1;
  for (int jt = 0; jt < ntiles; ++jt) {
    const int k0 = jt * 32;
    const _Float16* kp = kbuf + (head * kN + k0) * kDH;

    // ---- scores: S = q . k^T  (16 rows x 32 keys) ----
#pragma unroll
    for (int st = 0; st < 2; ++st) {
      v16h kb0 = load_fragB(kp + st * 16 * kDH,      kDH, lane);
      v16h kb1 = load_fragB(kp + st * 16 * kDH + 32, kDH, lane);
      v8f s = {};
      s = wmma_f16(qa0, kb0, s);
      s = wmma_f16(qa1, kb1, s);
      int col = st * 16 + ln;
#pragma unroll
      for (int r = 0; r < 8; ++r) Sld[r + 8 * hi][col] = s[r];
    }
    __syncthreads();

    // ---- online softmax with causal mask; lane pair (l, l+16) shares a row ----
    float pm = -1e30f;
#pragma unroll
    for (int c = 0; c < 16; ++c) {
      float xv = Sld[row][coff + c];
      if (k0 + coff + c <= g) pm = fmaxf(pm, xv);
    }
    pm = fmaxf(pm, __shfl_xor(pm, 16));
    float mnew  = fmaxf(Mrow, pm);
    float alpha = __expf(Mrow - mnew);
    float lsum = 0.0f;
#pragma unroll
    for (int c = 0; c < 16; ++c) {
      float p = (k0 + coff + c <= g) ? __expf(Sld[row][coff + c] - mnew) : 0.0f;
      Pld[row][coff + c] = (_Float16)p;
      lsum += p;
    }
    lsum += __shfl_xor(lsum, 16);
    Lrow = Lrow * alpha + lsum;
    Mrow = mnew;
    if (lane < 16) Ald[row] = alpha;
    __syncthreads();

    // ---- rescale accumulators ----
    float al[8];
#pragma unroll
    for (int r = 0; r < 8; ++r) al[r] = Ald[r + 8 * hi];
#pragma unroll
    for (int dt = 0; dt < 4; ++dt)
#pragma unroll
      for (int r = 0; r < 8; ++r) o[dt][r] *= al[r];

    // ---- O += P @ V  (K = 32 keys; V^T rows are contiguous over keys) ----
    v16h pa = load_fragA(&Pld[0][0], 32, lane);
    const _Float16* vp = vt + head * kDH * kN + k0;
#pragma unroll
    for (int dt = 0; dt < 4; ++dt) {
      v16h vb = load_fragB(vp + (size_t)dt * 16 * kN, kN, lane);
      o[dt] = wmma_f16(pa, vb, o[dt]);
    }
    __syncthreads();
  }

  if (lane < 16) Lld[lane] = Lrow;
  __syncthreads();
  float linv[8];
#pragma unroll
  for (int r = 0; r < 8; ++r) linv[r] = 1.0f / Lld[r + 8 * hi];

  // store sa as [B, N, H*DH]  (head concat)
#pragma unroll
  for (int dt = 0; dt < 4; ++dt)
#pragma unroll
    for (int r = 0; r < 8; ++r) {
      int rowg = r0 + r + 8 * hi;
      int dcol = h * kDH + dt * 16 + ln;
      sabuf[((size_t)b * kN + rowg) * kD + dcol] = (_Float16)(o[dt][r] * linv[r]);
    }
}

// ---------------------------------------------------------------------------
// Stage 3: out[b,n,i] = sum_j sa[b,n,j] * W_proj[i,j] + b_proj[i]
// One wave -> 16 rows x 64 cols. grid (N/16, D/64, B), block 32. Pipelined.
// ---------------------------------------------------------------------------
__global__ void out_proj(const _Float16* __restrict__ sa,
                         const _Float16* __restrict__ Wp,
                         const float* __restrict__ b_proj,
                         float* __restrict__ out) {
  const int lane = threadIdx.x & 31;
  const int hi   = lane >> 4;
  const int ln   = lane & 15;
  const int b = blockIdx.z;
  const int row0 = blockIdx.x * 16;
  const int d0   = blockIdx.y * 64;

  const _Float16* arow  = sa + ((size_t)b * kN + row0) * kD;
  const _Float16* wbase = Wp + (size_t)d0 * kD;   // W_proj rows = B columns

  v8f acc0 = {}, acc1 = {}, acc2 = {}, acc3 = {};
  v16h a  = load_fragA(arow, kD, lane);
  v16h b0 = load_fragB(wbase,           kD, lane);
  v16h b1 = load_fragB(wbase + 16 * kD, kD, lane);
  v16h b2 = load_fragB(wbase + 32 * kD, kD, lane);
  v16h b3 = load_fragB(wbase + 48 * kD, kD, lane);

#pragma unroll 1
  for (int kk = 0; kk < kD / 32 - 1; ++kk) {
    const int ko = (kk + 1) * 32;
    v16h an  = load_fragA(arow + ko, kD, lane);
    v16h bn0 = load_fragB(wbase + ko,           kD, lane);
    v16h bn1 = load_fragB(wbase + 16 * kD + ko, kD, lane);
    v16h bn2 = load_fragB(wbase + 32 * kD + ko, kD, lane);
    v16h bn3 = load_fragB(wbase + 48 * kD + ko, kD, lane);
    acc0 = wmma_f16(a, b0, acc0);
    acc1 = wmma_f16(a, b1, acc1);
    acc2 = wmma_f16(a, b2, acc2);
    acc3 = wmma_f16(a, b3, acc3);
    a = an; b0 = bn0; b1 = bn1; b2 = bn2; b3 = bn3;
  }
  acc0 = wmma_f16(a, b0, acc0);
  acc1 = wmma_f16(a, b1, acc1);
  acc2 = wmma_f16(a, b2, acc2);
  acc3 = wmma_f16(a, b3, acc3);
  v8f acc[4] = {acc0, acc1, acc2, acc3};

#pragma unroll
  for (int dt = 0; dt < 4; ++dt) {
    int dcol = d0 + dt * 16 + ln;
    float bv = b_proj[dcol];
#pragma unroll
    for (int r = 0; r < 8; ++r) {
      int row = row0 + r + 8 * hi;
      out[((size_t)b * kN + row) * kD + dcol] = acc[dt][r] + bv;
    }
  }
}

// ---------------------------------------------------------------------------
extern "C" void kernel_launch(void* const* d_in, const int* in_sizes, int n_in,
                              void* d_out, int out_size, void* d_ws, size_t ws_size,
                              hipStream_t stream) {
  (void)in_sizes; (void)n_in; (void)out_size; (void)ws_size;
  const float* x      = (const float*)d_in[0];
  const float* W_kqv  = (const float*)d_in[1];
  const float* b_kqv  = (const float*)d_in[2];
  const float* W_proj = (const float*)d_in[3];
  const float* b_proj = (const float*)d_in[4];
  float* out = (float*)d_out;

  char* ws = (char*)d_ws;
  size_t off = 0;
  auto carve = [&](size_t elems) -> _Float16* {
    _Float16* p = (_Float16*)(ws + off);
    off += ((elems * sizeof(_Float16)) + 255) & ~(size_t)255;
    return p;
  };
  _Float16* xh   = carve((size_t)kB * kN * kD);        // x in f16
  _Float16* Wt   = carve((size_t)kH * kE * kD);        // W_kqv^T per head
  _Float16* Wp   = carve((size_t)kD * kD);             // W_proj f16
  _Float16* qbuf = carve((size_t)kB * kH * kN * kDH);  // q (pre-scaled) [b,h,n,d]
  _Float16* kbuf = carve((size_t)kB * kH * kN * kDH);  // k              [b,h,n,d]
  _Float16* vt   = carve((size_t)kB * kH * kDH * kN);  // v transposed   [b,h,d,n]
  _Float16* sab  = carve((size_t)kB * kN * kD);        // attention output

  cvt_f32_to_f16<<<1024, 256, 0, stream>>>(x, xh, kB * kN * kD);
  cvt_wkqv_t    <<<1024, 256, 0, stream>>>(W_kqv, Wt);
  cvt_f32_to_f16<<<1024, 256, 0, stream>>>(W_proj, Wp, kD * kD);

  kqv_gemm  <<<dim3(kN / 16, 3, kB * kH),  32, 0, stream>>>(xh, Wt, b_kqv,
                                                            qbuf, kbuf, vt);
  flash_attn<<<dim3(kN / 16, kB * kH),     32, 0, stream>>>(qbuf, kbuf, vt, sab);
  out_proj  <<<dim3(kN / 16, kD / 64, kB), 32, 0, stream>>>(sab, Wp, b_proj, out);
}
